// MomGRUAddingBaseline_87351044866256
// MI455X (gfx1250) — matrix operand order repaired and checked by reading, not compiled, not run on hardware
//
#include <hip/hip_runtime.h>

typedef __attribute__((ext_vector_type(16))) __bf16 v16bf;
typedef __attribute__((ext_vector_type(8)))  float  v8f;

#define H_   256
#define T_   1024
#define B_   128
#define NWG  16
#define NTHR 256
#define WSTR 264              // padded LDS row stride (bf16 elems): bank-conflict-free 16B reads
#define NSLOT 1024            // flag slots (steps 0..1022 use 0..1022; last step uses counter)

__device__ __forceinline__ float sigm(float v)  { return 1.f / (1.f + __expf(-v)); }
__device__ __forceinline__ float tanhf_(float v){ float e = __expf(-2.f * v); return (1.f - e) / (1.f + e); }

// ---------------- prelude: zero h buffers (buffer 0) + sync flags + exit counter -------------
__global__ __launch_bounds__(NTHR) void momgru_init(float* __restrict__ hf0,
                                                    unsigned* __restrict__ hb0_u32,
                                                    unsigned* __restrict__ flags) {
  int gt = blockIdx.x * blockDim.x + threadIdx.x;
  int stride = gridDim.x * blockDim.x;
  for (int i = gt; i < B_ * H_; i += stride)            hf0[i] = 0.f;       // f32 h, buffer 0
  for (int i = gt; i < (B_ * H_) / 2; i += stride)      hb0_u32[i] = 0u;    // bf16 h, buffer 0
  for (int i = gt; i < NSLOT * NWG + 1; i += stride)    flags[i] = 0u;      // flags + exit counter
}

// ---------------- grid barrier helpers ----------------
__device__ __forceinline__ void flag_sync(unsigned* flags, int s, int wg) {
  __threadfence();
  __syncthreads();
  if (threadIdx.x == 0) {
    __hip_atomic_store(&flags[s * NWG + wg], 1u, __ATOMIC_RELEASE, __HIP_MEMORY_SCOPE_AGENT);
    for (int j = 0; j < NWG; ++j) {
      while (__hip_atomic_load(&flags[s * NWG + j], __ATOMIC_ACQUIRE, __HIP_MEMORY_SCOPE_AGENT) != 1u)
        __builtin_amdgcn_s_sleep(1);
    }
  }
  __syncthreads();
  __threadfence();
}

__device__ __forceinline__ void exit_sync(unsigned* ec) {
  __threadfence();
  __syncthreads();
  if (threadIdx.x == 0) {
    __hip_atomic_fetch_add(ec, 1u, __ATOMIC_ACQ_REL, __HIP_MEMORY_SCOPE_AGENT);
    while (__hip_atomic_load(ec, __ATOMIC_ACQUIRE, __HIP_MEMORY_SCOPE_AGENT) < (unsigned)NWG)
      __builtin_amdgcn_s_sleep(1);
  }
  __syncthreads();
  __threadfence();
}

// ---------------- main persistent kernel ----------------
__global__ __launch_bounds__(NTHR, 1)
void momgru_main(const float* __restrict__ x,      // [B,T,2]
                 const float* __restrict__ Wih,    // [768,2]
                 const float* __restrict__ Whh,    // [768,256]
                 const float* __restrict__ bih,    // [768]
                 const float* __restrict__ bhh,    // [768]
                 const float* __restrict__ Wbx,    // [2]
                 const float* __restrict__ Wbh,    // [256]
                 const float* __restrict__ bbeta_p,// [1]
                 const float* __restrict__ s_p,    // [1]
                 const float* __restrict__ Whead,  // [256]
                 const float* __restrict__ bhead_p,// [1]
                 float* __restrict__ out,          // [128] head, then [B,T] betas
                 float* __restrict__ hf32,         // 2 x [B,H] f32 state
                 __bf16* __restrict__ hbf,         // 2 x [B,H] bf16 state
                 unsigned* __restrict__ flags) {   // NSLOT*NWG flags + 1 exit counter
  unsigned* ec = flags + NSLOT * NWG;

  const int wg   = blockIdx.x;
  const int tid  = threadIdx.x;
  const int w    = tid >> 5;           // wave id (8 waves) -> batch rows 16w..16w+15
  const int l    = tid & 31;
  const int lc   = l & 15;
  const int half = l >> 4;
  const int row_m = 16 * w + lc;       // A-fragment row / beta row for this lane
  const int colh  = 16 * wg + lc;      // this WG's hidden column owned by this lane

  // --- LDS: W_hh slice (48 rows x 256, bf16, padded) + Wb_h ---
  __shared__ __align__(16) __bf16 wlds[48 * WSTR];
  __shared__ __align__(16) float  wbh_s[H_];

  for (int e = tid; e < 48 * 256; e += NTHR) {
    int rr = e >> 8, c = e & 255;
    int g = rr >> 4, cl = rr & 15;
    int grow = g * 256 + wg * 16 + cl;             // global W_hh row (gate-interleaved ownership)
    wlds[rr * WSTR + c] = (__bf16)Whh[grow * 256 + c];
  }
  if (tid < H_) wbh_s[tid] = Wbh[tid];

  // --- per-lane constants for owned columns (one per gate) ---
  float wih0[3], wih1[3], bihv[3], bhhv[3];
#pragma unroll
  for (int g = 0; g < 3; ++g) {
    int cg = g * 256 + wg * 16 + lc;
    wih0[g] = Wih[cg * 2 + 0];
    wih1[g] = Wih[cg * 2 + 1];
    bihv[g] = bih[cg];
    bhhv[g] = bhh[cg];
  }
  const float sv = s_p[0], wbx0 = Wbx[0], wbx1 = Wbx[1], bbeta = bbeta_p[0], bhead = bhead_p[0];
  __syncthreads();

  // --- register-resident recurrent state ---
  float vst[3][8];                                  // momentum buffer slice (C/D layout)
  float hprev[8];                                   // own h(t) entries for z*h term
#pragma unroll
  for (int j = 0; j < 8; ++j) { hprev[j] = 0.f; vst[0][j] = 0.f; vst[1][j] = 0.f; vst[2][j] = 0.f; }

  const float2* x2 = (const float2*)x;

  for (int t = 0; t < T_; ++t) {
    const int cur = t & 1, nxt = cur ^ 1;
    const float*  hf   = hf32 + cur * B_ * H_;
    const __bf16* hb   = hbf  + cur * B_ * H_;
    float*        hf_n = hf32 + nxt * B_ * H_;
    __bf16*       hb_n = hbf  + nxt * B_ * H_;

    // ---- beta_t = sigmoid(x_t.Wbx + h.Wbh + b) : half-wave split dot, shfl_xor reduce ----
    float bsum = 0.f;
    {
      const float4* hr4 = (const float4*)(hf + row_m * H_ + half * 128);
      const float4* wb4 = (const float4*)(wbh_s + half * 128);
#pragma unroll
      for (int i = 0; i < 32; ++i) {
        float4 a = hr4[i], b = wb4[i];
        bsum = fmaf(a.x, b.x, fmaf(a.y, b.y, fmaf(a.z, b.z, fmaf(a.w, b.w, bsum))));
      }
    }
    bsum += __shfl_xor(bsum, 16);
    float2 xr = x2[row_m * T_ + t];
    float beta_row = sigm(bsum + xr.x * wbx0 + xr.y * wbx1 + bbeta);
    if (wg == 0 && l < 16) out[B_ + row_m * T_ + t] = beta_row;   // betas[B,T]

    // ---- gh tile = h @ Whh_slice^T + b_hh via WMMA bf16 (3 gate tiles, K=256) ----
    v8f acc[3];
#pragma unroll
    for (int g = 0; g < 3; ++g)
#pragma unroll
      for (int j = 0; j < 8; ++j) acc[g][j] = bhhv[g];

    const __bf16* arow = hb + row_m * H_;
#pragma unroll
    for (int kb = 0; kb < 8; ++kb) {
      const int kbase = kb * 32;
      union { v16bf v; uint4 q[2]; } A, Bf;
      A.q[0] = *(const uint4*)(arow + kbase + 8 * half);          // K = 8h .. 8h+7
      A.q[1] = *(const uint4*)(arow + kbase + 16 + 8 * half);     // K = 16+8h .. 16+8h+7
#pragma unroll
      for (int g = 0; g < 3; ++g) {
        const __bf16* bl = &wlds[(g * 16 + lc) * WSTR + kbase + 8 * half];
        Bf.q[0] = *(const uint4*)(bl);                            // K = 8h .. 8h+7
        Bf.q[1] = *(const uint4*)(bl + 16);                       // K = 16+8h .. 16+8h+7
        acc[g] = __builtin_amdgcn_wmma_f32_16x16x32_bf16(
            false, A.v, false, Bf.v, (short)0, acc[g], false, false);
      }
    }

    // ---- momentum + gates + state update (register layout == WMMA C/D layout) ----
#pragma unroll
    for (int j = 0; j < 8; ++j) {
      int br = 16 * w + 8 * half + j;
      float2 xj = x2[br * T_ + t];
      float bj = __shfl(beta_row, 8 * half + j);
      float pr = fmaf(xj.x, wih0[0], fmaf(xj.y, wih1[0], bihv[0]));
      float pz = fmaf(xj.x, wih0[1], fmaf(xj.y, wih1[1], bihv[1]));
      float pn = fmaf(xj.x, wih0[2], fmaf(xj.y, wih1[2], bihv[2]));
      vst[0][j] = fmaf(bj, vst[0][j], sv * pr);
      vst[1][j] = fmaf(bj, vst[1][j], sv * pz);
      vst[2][j] = fmaf(bj, vst[2][j], sv * pn);
      float r  = sigm(vst[0][j] + acc[0][j]);
      float z  = sigm(vst[1][j] + acc[1][j]);
      float n  = tanhf_(vst[2][j] + r * acc[2][j]);
      float hn = (1.f - z) * n + z * hprev[j];
      hprev[j] = hn;
      hf_n[br * H_ + colh] = hn;
      hb_n[br * H_ + colh] = (__bf16)hn;
    }

    // ---- device-wide step barrier ----
    if (t < T_ - 1) flag_sync(flags, t, wg);
    else            exit_sync(ec);
  }

  // ---- read-out head on h_T (buffer 0, since T is even) ----
  if (wg == 0) {
    float hsum = 0.f;
    const float4* h4 = (const float4*)(hf32 + row_m * H_ + half * 128);
    const float4* w4 = (const float4*)(Whead + half * 128);
#pragma unroll
    for (int i = 0; i < 32; ++i) {
      float4 a = h4[i], b = w4[i];
      hsum = fmaf(a.x, b.x, fmaf(a.y, b.y, fmaf(a.z, b.z, fmaf(a.w, b.w, hsum))));
    }
    hsum += __shfl_xor(hsum, 16);
    if (l < 16) out[row_m] = hsum + bhead;
  }
}

extern "C" void kernel_launch(void* const* d_in, const int* in_sizes, int n_in,
                              void* d_out, int out_size, void* d_ws, size_t ws_size,
                              hipStream_t stream) {
  const float* x      = (const float*)d_in[0];
  const float* Wih    = (const float*)d_in[1];
  const float* Whh    = (const float*)d_in[2];
  const float* bih    = (const float*)d_in[3];
  const float* bhh    = (const float*)d_in[4];
  const float* Wbx    = (const float*)d_in[5];
  const float* Wbh    = (const float*)d_in[6];
  const float* bbeta  = (const float*)d_in[7];
  const float* s_p    = (const float*)d_in[8];
  const float* Whead  = (const float*)d_in[9];
  const float* bhead  = (const float*)d_in[10];
  float* out = (float*)d_out;

  char* ws = (char*)d_ws;
  float*    hf32  = (float*)ws;                                   // 2*128*256*4 = 262144 B
  __bf16*   hbf   = (__bf16*)(ws + 262144);                       // 2*128*256*2 = 131072 B
  unsigned* flags = (unsigned*)(ws + 262144 + 131072);            // (1024*16+1)*4 B

  momgru_init<<<dim3(64), dim3(NTHR), 0, stream>>>(hf32, (unsigned*)hbf, flags);
  momgru_main<<<dim3(NWG), dim3(NTHR), 0, stream>>>(
      x, Wih, Whh, bih, bhh, Wbx, Wbh, bbeta, s_p, Whead, bhead,
      out, hf32, hbf, flags);
}
